// GraphNeuralNetworkEncoder_49220325212176
// MI455X (gfx1250) — matrix operand order repaired and checked by reading, not compiled, not run on hardware
//
#include <hip/hip_runtime.h>

// ---------------------------------------------------------------------------
// GNN encoder on gfx1250 (MI455X):
//   edge MLP + scatter-add via v_wmma_f32_16x16x32_bf16 + global_atomic_add_f32
//   node MLP via v_wmma_f32_16x16x32_bf16
// M-blocked: each wave handles 32 rows (2 WMMA M-tiles), so every B-fragment
// load feeds TWO WMMAs -> halves weight-fetch per FLOP vs 16-row tiling.
// ---------------------------------------------------------------------------

typedef __attribute__((ext_vector_type(16))) __bf16 v16bf;
typedef __attribute__((ext_vector_type(8)))  __bf16 v8bf;
typedef __attribute__((ext_vector_type(8)))  float  v8f;

#define NN 50000
#define NE 625000
#define DD 128

union AFrag { v16bf v; v8bf h[2]; };

__device__ __forceinline__ v8f wmma_bf16(v16bf a, v16bf b, v8f c) {
  return __builtin_amdgcn_wmma_f32_16x16x32_bf16(false, a, false, b, (short)0, c,
                                                 false, false);
}

// ------------------------------ prep kernels -------------------------------

__global__ void zero_f32_kernel(float4* __restrict__ p, int n4) {
  int i = blockIdx.x * blockDim.x + threadIdx.x;
  if (i < n4) p[i] = make_float4(0.f, 0.f, 0.f, 0.f);
}

__global__ void cvt_f32_bf16_kernel(const float* __restrict__ in,
                                    __bf16* __restrict__ out, int n) {
  int i = blockIdx.x * blockDim.x + threadIdx.x;
  if (i < n) out[i] = (__bf16)in[i];
}

// Pack a row-major [K, Ndim] f32 weight matrix into per-lane B-fragment order:
// packed[((kc*NT + nt)*32 + lane)*16 + e] = W[kc*32 + (e>=8?16:0) + (lane>=16?8:0) + e%8][nt*16 + lane%16]
// so each wave fetches a B tile as one contiguous 32B run per lane.
__global__ void pack_w_kernel(const float* __restrict__ W,
                              __bf16* __restrict__ out, int Ndim, int total) {
  int idx = blockIdx.x * blockDim.x + threadIdx.x;
  if (idx >= total) return;
  int e    = idx & 15;
  int l    = (idx >> 4) & 31;
  int rest = idx >> 9;
  int NT   = Ndim >> 4;
  int nt   = rest % NT;
  int kc   = rest / NT;
  int K    = kc * 32 + ((e >> 3) << 4) + ((l >> 4) << 3) + (e & 7);
  int n    = nt * 16 + (l & 15);
  out[idx] = (__bf16)W[K * Ndim + n];
}

// ------------------------------ edge kernel --------------------------------
// One wave = 32 edges (2 M-tiles). Layer1: [32,256]x[256,128]
// (8 kc x 8 nt x 2 mt WMMAs), ReLU -> bf16 LDS tile,
// Layer2: [32,128]x[128,128], then global_atomic_add_f32 scatter.
__global__ __launch_bounds__(128) void edge_msg_kernel(
    const __bf16* __restrict__ nodeB,
    const long long* __restrict__ eidx,
    const float* __restrict__ b1, const float* __restrict__ b2,
    const __bf16* __restrict__ w1p, const __bf16* __restrict__ w2p,
    float* __restrict__ agg) {
  __shared__ __align__(16) __bf16 h1[4][32][136];  // 136 stride staggers banks
  const int lane = threadIdx.x & 31;
  const int w    = threadIdx.x >> 5;
  const long long e0 = (long long)(blockIdx.x * 4 + w) * 32;
  if (e0 >= NE) return;
  const int col  = lane & 15;
  const int half = lane >> 4;
  const int ho   = half * 8;  // A-layout: lanes>=16 hold the +8 K sub-block

  const long long* src = eidx;
  const long long* tgt = eidx + NE;

  const __bf16* srow[2];
  const __bf16* trow[2];
#pragma unroll
  for (int mt = 0; mt < 2; ++mt) {
    long long eg = e0 + mt * 16 + col;
    if (eg >= NE) eg = NE - 1;  // clamp gather; scatter guarded below
    srow[mt] = nodeB + (long long)src[eg] * DD;
    trow[mt] = nodeB + (long long)tgt[eg] * DD;
  }

  // ---- layer 1 ----
  v8f acc[2][8];
#pragma unroll
  for (int nt = 0; nt < 8; ++nt) {
    float bv = b1[nt * 16 + col];
#pragma unroll
    for (int mt = 0; mt < 2; ++mt)
#pragma unroll
      for (int r = 0; r < 8; ++r) acc[mt][nt][r] = bv;
  }
#pragma unroll
  for (int kc = 0; kc < 8; ++kc) {
    AFrag a[2];
#pragma unroll
    for (int mt = 0; mt < 2; ++mt) {
      const __bf16* base =
          (kc < 4) ? (srow[mt] + kc * 32) : (trow[mt] + (kc - 4) * 32);
      a[mt].h[0] = *(const v8bf*)(base + ho);
      a[mt].h[1] = *(const v8bf*)(base + 16 + ho);
    }
#pragma unroll
    for (int nt = 0; nt < 8; ++nt) {
      v16bf b = *(const v16bf*)(w1p + ((kc * 8 + nt) * 512 + lane * 16));
      acc[0][nt] = wmma_bf16(a[0].v, b, acc[0][nt]);  // one B, two WMMAs
      acc[1][nt] = wmma_bf16(a[1].v, b, acc[1][nt]);
    }
  }
  // ReLU -> bf16 LDS (C layout: row = r + 8*half, col = nt*16 + col)
#pragma unroll
  for (int mt = 0; mt < 2; ++mt)
#pragma unroll
    for (int nt = 0; nt < 8; ++nt)
#pragma unroll
      for (int r = 0; r < 8; ++r) {
        float v = acc[mt][nt][r];
        v = v > 0.f ? v : 0.f;
        h1[w][mt * 16 + r + half * 8][nt * 16 + col] = (__bf16)v;
      }

  // ---- layer 2 ----
  v8f acc2[2][8];
#pragma unroll
  for (int nt = 0; nt < 8; ++nt) {
    float bv = b2[nt * 16 + col];
#pragma unroll
    for (int mt = 0; mt < 2; ++mt)
#pragma unroll
      for (int r = 0; r < 8; ++r) acc2[mt][nt][r] = bv;
  }
  const __bf16* lr[2] = {&h1[w][col][0], &h1[w][16 + col][0]};
#pragma unroll
  for (int kc = 0; kc < 4; ++kc) {
    AFrag a[2];
#pragma unroll
    for (int mt = 0; mt < 2; ++mt) {
      a[mt].h[0] = *(const v8bf*)(lr[mt] + kc * 32 + ho);
      a[mt].h[1] = *(const v8bf*)(lr[mt] + kc * 32 + 16 + ho);
    }
#pragma unroll
    for (int nt = 0; nt < 8; ++nt) {
      v16bf b = *(const v16bf*)(w2p + ((kc * 8 + nt) * 512 + lane * 16));
      acc2[0][nt] = wmma_bf16(a[0].v, b, acc2[0][nt]);
      acc2[1][nt] = wmma_bf16(a[1].v, b, acc2[1][nt]);
    }
  }

  // ---- scatter-add (hardware f32 atomics, no-return) ----
#pragma unroll
  for (int mt = 0; mt < 2; ++mt) {
    long long trows[8];
#pragma unroll
    for (int r = 0; r < 8; ++r) {
      long long em = e0 + mt * 16 + r + half * 8;
      trows[r] = (em < NE) ? tgt[em] : -1;
    }
#pragma unroll
    for (int r = 0; r < 8; ++r) {
      if (trows[r] >= 0) {
        float* dst = agg + trows[r] * DD;
#pragma unroll
        for (int nt = 0; nt < 8; ++nt)
          unsafeAtomicAdd(dst + nt * 16 + col, acc2[mt][nt][r]);
      }
    }
  }
}

// ------------------------------ node kernel --------------------------------
__global__ __launch_bounds__(128) void node_update_kernel(
    const __bf16* __restrict__ nodeB, const __bf16* __restrict__ aggB,
    const float* __restrict__ b1, const float* __restrict__ b2,
    const __bf16* __restrict__ w1p, const __bf16* __restrict__ w2p,
    float* __restrict__ out) {
  __shared__ __align__(16) __bf16 h1[4][32][136];
  const int lane = threadIdx.x & 31;
  const int w    = threadIdx.x >> 5;
  const int n0   = (blockIdx.x * 4 + w) * 32;
  if (n0 >= NN) return;
  const int col  = lane & 15;
  const int half = lane >> 4;
  const int ho   = half * 8;

  const __bf16* srow[2];
  const __bf16* trow[2];
#pragma unroll
  for (int mt = 0; mt < 2; ++mt) {
    int n = n0 + mt * 16 + col;
    if (n >= NN) n = NN - 1;  // clamp gather; stores guarded below
    srow[mt] = nodeB + (long long)n * DD;
    trow[mt] = aggB + (long long)n * DD;
  }

  // ---- layer 1: [32,256] x [256,128] ----
  v8f acc[2][8];
#pragma unroll
  for (int nt = 0; nt < 8; ++nt) {
    float bv = b1[nt * 16 + col];
#pragma unroll
    for (int mt = 0; mt < 2; ++mt)
#pragma unroll
      for (int r = 0; r < 8; ++r) acc[mt][nt][r] = bv;
  }
#pragma unroll
  for (int kc = 0; kc < 8; ++kc) {
    AFrag a[2];
#pragma unroll
    for (int mt = 0; mt < 2; ++mt) {
      const __bf16* base =
          (kc < 4) ? (srow[mt] + kc * 32) : (trow[mt] + (kc - 4) * 32);
      a[mt].h[0] = *(const v8bf*)(base + ho);
      a[mt].h[1] = *(const v8bf*)(base + 16 + ho);
    }
#pragma unroll
    for (int nt = 0; nt < 8; ++nt) {
      v16bf b = *(const v16bf*)(w1p + ((kc * 8 + nt) * 512 + lane * 16));
      acc[0][nt] = wmma_bf16(a[0].v, b, acc[0][nt]);
      acc[1][nt] = wmma_bf16(a[1].v, b, acc[1][nt]);
    }
  }
#pragma unroll
  for (int mt = 0; mt < 2; ++mt)
#pragma unroll
    for (int nt = 0; nt < 8; ++nt)
#pragma unroll
      for (int r = 0; r < 8; ++r) {
        float v = acc[mt][nt][r];
        v = v > 0.f ? v : 0.f;
        h1[w][mt * 16 + r + half * 8][nt * 16 + col] = (__bf16)v;
      }

  // ---- layer 2: [32,128] x [128,64] ----
  v8f acc2[2][4];
#pragma unroll
  for (int nt = 0; nt < 4; ++nt) {
    float bv = b2[nt * 16 + col];
#pragma unroll
    for (int mt = 0; mt < 2; ++mt)
#pragma unroll
      for (int r = 0; r < 8; ++r) acc2[mt][nt][r] = bv;
  }
  const __bf16* lr[2] = {&h1[w][col][0], &h1[w][16 + col][0]};
#pragma unroll
  for (int kc = 0; kc < 4; ++kc) {
    AFrag a[2];
#pragma unroll
    for (int mt = 0; mt < 2; ++mt) {
      a[mt].h[0] = *(const v8bf*)(lr[mt] + kc * 32 + ho);
      a[mt].h[1] = *(const v8bf*)(lr[mt] + kc * 32 + 16 + ho);
    }
#pragma unroll
    for (int nt = 0; nt < 4; ++nt) {
      v16bf b = *(const v16bf*)(w2p + ((kc * 4 + nt) * 512 + lane * 16));
      acc2[0][nt] = wmma_bf16(a[0].v, b, acc2[0][nt]);
      acc2[1][nt] = wmma_bf16(a[1].v, b, acc2[1][nt]);
    }
  }
#pragma unroll
  for (int mt = 0; mt < 2; ++mt)
#pragma unroll
    for (int nt = 0; nt < 4; ++nt)
#pragma unroll
      for (int r = 0; r < 8; ++r) {
        int n = n0 + mt * 16 + r + half * 8;
        if (n < NN) out[(long long)n * 64 + nt * 16 + col] = acc2[mt][nt][r];
      }
}

// ------------------------------ launcher -----------------------------------

extern "C" void kernel_launch(void* const* d_in, const int* in_sizes, int n_in,
                              void* d_out, int out_size, void* d_ws,
                              size_t ws_size, hipStream_t stream) {
  const float*     nodeF  = (const float*)d_in[0];
  const long long* eidx   = (const long long*)d_in[1];
  const float*     msg_w1 = (const float*)d_in[2];
  const float*     msg_b1 = (const float*)d_in[3];
  const float*     msg_w2 = (const float*)d_in[4];
  const float*     msg_b2 = (const float*)d_in[5];
  const float*     upd_w1 = (const float*)d_in[6];
  const float*     upd_b1 = (const float*)d_in[7];
  const float*     upd_w2 = (const float*)d_in[8];
  const float*     upd_b2 = (const float*)d_in[9];
  float* out = (float*)d_out;

  // workspace layout (256B-aligned offsets)
  char* ws = (char*)d_ws;
  float*  agg   = (float*)(ws);                  // NN*128 f32  = 25,600,000 B
  __bf16* nodeB = (__bf16*)(ws + 25600000);      // NN*128 bf16 = 12,800,000 B
  __bf16* aggB  = (__bf16*)(ws + 38400000);      // NN*128 bf16 = 12,800,000 B
  __bf16* w1m   = (__bf16*)(ws + 51200000);      // 256*128 bf16
  __bf16* w2m   = (__bf16*)(ws + 51265536);      // 128*128 bf16
  __bf16* w1u   = (__bf16*)(ws + 51298304);      // 256*128 bf16
  __bf16* w2u   = (__bf16*)(ws + 51363840);      // 128*64 bf16

  // prep
  int n4 = NN * DD / 4;
  zero_f32_kernel<<<(n4 + 255) / 256, 256, 0, stream>>>((float4*)agg, n4);
  cvt_f32_bf16_kernel<<<(NN * DD + 255) / 256, 256, 0, stream>>>(nodeF, nodeB,
                                                                 NN * DD);
  pack_w_kernel<<<(256 * 128 + 255) / 256, 256, 0, stream>>>(msg_w1, w1m, 128,
                                                             256 * 128);
  pack_w_kernel<<<(128 * 128 + 255) / 256, 256, 0, stream>>>(msg_w2, w2m, 128,
                                                             128 * 128);
  pack_w_kernel<<<(256 * 128 + 255) / 256, 256, 0, stream>>>(upd_w1, w1u, 128,
                                                             256 * 128);
  pack_w_kernel<<<(128 * 64 + 255) / 256, 256, 0, stream>>>(upd_w2, w2u, 64,
                                                            128 * 64);

  // edge MLP + scatter: 4 waves/block, 32 edges/wave
  int etiles = (NE + 31) / 32;
  edge_msg_kernel<<<(etiles + 3) / 4, 128, 0, stream>>>(
      nodeB, eidx, msg_b1, msg_b2, w1m, w2m, agg);

  // aggregate -> bf16, then node MLP
  cvt_f32_bf16_kernel<<<(NN * DD + 255) / 256, 256, 0, stream>>>(agg, aggB,
                                                                 NN * DD);
  int ntiles = (NN + 31) / 32;
  node_update_kernel<<<(ntiles + 3) / 4, 128, 0, stream>>>(
      nodeB, aggB, upd_b1, upd_b2, w1u, w2u, out);
}
